// NewsEncoder_13889924235612
// MI455X (gfx1250) — compile-verified, matrix operand order checked by reading, not run on hardware
//
#include <hip/hip_runtime.h>
#include <math.h>

typedef _Float16 h16;
typedef unsigned int u32;
typedef __attribute__((ext_vector_type(16))) _Float16 v16h;
typedef __attribute__((ext_vector_type(8)))  _Float16 v8h;
typedef __attribute__((ext_vector_type(8)))  float    v8f;
typedef __attribute__((ext_vector_type(4)))  u32      v4u;
typedef __attribute__((ext_vector_type(8)))  int      v8i;
typedef __attribute__((ext_vector_type(4)))  int      v4i;

#define BM 128
#define BN 128
#define BKK 32
#define LDT 40   // LDS row stride in halves (80B): 64B data row + 16B TDM pad

#if defined(__has_builtin)
#  if __has_builtin(__builtin_amdgcn_tensor_load_to_lds)
#    define USE_TDM 1
#  endif
#endif
#ifndef USE_TDM
#  define USE_TDM 0
#endif

// ---------------------------------------------------------------------------
// TDM tile load: 2D tile (32 halves x 128 rows) from row-major [rows x K] f16
// into LDS with 16B pad after each 64B row (=> LDS stride LDT=40 halves).
// D# packing per CDNA5 ISA 8.3/8.4. Issued once per wave; EXEC-independent,
// so callers gate it with a wave-0 branch. Tracked by TENSORcnt.
// ---------------------------------------------------------------------------
#if USE_TDM
__device__ __forceinline__ void tdm_tile(const h16* gptr, u32 lds_off,
                                         int K, int rows) {
  unsigned long long ga = (unsigned long long)(size_t)gptr;
  v4u g0;
  g0[0] = 1u;                                   // count=1, user mode
  g0[1] = lds_off;                              // lds_addr (bytes)
  g0[2] = (u32)(ga & 0xffffffffu);              // global_addr[31:0]
  g0[3] = (u32)((ga >> 32) & 0x01ffffffu)       // global_addr[56:32]
          | (2u << 30);                         // type=2 (image)
  v8i g1;
  g1[0] = (int)((1u << 16)                      // data_size=1 -> 2 bytes
                | (1u << 20)                    // pad_enable
                | (3u << 22)                    // pad_interval: 16 dwords (64B)
                | (3u << 25));                  // pad_amount:   4 dwords (16B)
  g1[1] = (int)(((u32)K & 0xffffu) << 16);      // tensor_dim0[15:0]
  g1[2] = (int)((((u32)K >> 16) & 0xffffu)      // tensor_dim0[31:16]
                | (((u32)rows & 0xffffu) << 16));// tensor_dim1[15:0]
  g1[3] = (int)((((u32)rows >> 16) & 0xffffu)   // tensor_dim1[31:16]
                | ((u32)BKK << 16));            // tile_dim0 = 32 elems
  g1[4] = (int)(u32)BM;                         // tile_dim1 = 128, tile_dim2=0
  g1[5] = (int)(u32)K;                          // tensor_dim0_stride[31:0]
  g1[6] = 0;                                    // stride0[47:32], stride1 lo
  g1[7] = 0;
  v4i z4 = {0, 0, 0, 0};
#if __clang_major__ >= 23
  v8i z8 = {0, 0, 0, 0, 0, 0, 0, 0};
  __builtin_amdgcn_tensor_load_to_lds(g0, g1, z4, z4, z8, 0);
#else
  __builtin_amdgcn_tensor_load_to_lds(g0, g1, z4, z4, 0);
#endif
}
#endif

// ---------------------------------------------------------------------------
// Fragment load from LDS matching CDNA5 16-bit A/B WMMA layout:
// lanes 0-15 hold K {0..7, 16..23}, lanes 16-31 hold K {8..15, 24..31}.
// ---------------------------------------------------------------------------
__device__ __forceinline__ v16h frag_ld(const h16* p) {
  v8h lo = *(const v8h*)(p);
  v8h hi = *(const v8h*)(p + 16);
  v16h r;
#pragma unroll
  for (int i = 0; i < 8; ++i) { r[i] = lo[i]; r[i + 8] = hi[i]; }
  return r;
}

// ---------------------------------------------------------------------------
// GEMM: out[M,N] = A[M,K] (f16) * W[K,N] (given transposed f16 Bt[N,K]) + bias
// EPI: 0 bias->f16 | 1 bias+GELU->f16 | 2 bias+resid->f32 (aliasable) | 3 ->f32
// 256 thr (8 waves), 128x128 tile, BK=32; wave grid 2x4, 4x2 WMMA per wave.
// TDM double-buffered LDS staging; wave0 drives the tensor pipeline.
// ---------------------------------------------------------------------------
template<int EPI>
__global__ __launch_bounds__(256) void gemm16(
    const h16* __restrict__ A, const h16* __restrict__ Bt,
    const float* __restrict__ bias, void* __restrict__ outp,
    const float* __restrict__ resid, int M, int N, int K) {
#if USE_TDM
  __shared__ h16 As[2][BM * LDT];
  __shared__ h16 Bs[2][BN * LDT];
#else
  __shared__ h16 As[1][BM * LDT];
  __shared__ h16 Bs[1][BN * LDT];
#endif
  const int tid  = threadIdx.x;
  const int lane = tid & 31;
  const int wave = tid >> 5;
  const int wm = (wave & 1) * 64;
  const int wn = (wave >> 1) * 32;
  const int m0 = blockIdx.x * BM;
  const int n0 = blockIdx.y * BN;
  const int nk = K / BKK;

  v8f acc[4][2] = {};

#if USE_TDM
  const h16* Ab = A  + (size_t)m0 * K;
  const h16* Bb = Bt + (size_t)n0 * K;
  if (wave == 0) {
    tdm_tile(Ab, (u32)(size_t)(void*)&As[0][0], K, M);
    tdm_tile(Bb, (u32)(size_t)(void*)&Bs[0][0], K, N);
  }
  for (int kt = 0; kt < nk; ++kt) {
    const int cur = kt & 1;
    if (wave == 0) {
      if (kt + 1 < nk) {
        tdm_tile(Ab + (size_t)(kt + 1) * BKK,
                 (u32)(size_t)(void*)&As[cur ^ 1][0], K, M);
        tdm_tile(Bb + (size_t)(kt + 1) * BKK,
                 (u32)(size_t)(void*)&Bs[cur ^ 1][0], K, N);
        __builtin_amdgcn_s_wait_tensorcnt(2);  // current pair retired
      } else {
        __builtin_amdgcn_s_wait_tensorcnt(0);  // drain
      }
    }
    __syncthreads();
    const int lr = lane & 15;
    const int lh = (lane >> 4) * 8;
    v16h af[4], bf[2];
#pragma unroll
    for (int i = 0; i < 4; ++i)
      af[i] = frag_ld(&As[cur][(wm + i * 16 + lr) * LDT + lh]);
#pragma unroll
    for (int j = 0; j < 2; ++j)
      bf[j] = frag_ld(&Bs[cur][(wn + j * 16 + lr) * LDT + lh]);
#pragma unroll
    for (int i = 0; i < 4; ++i)
#pragma unroll
      for (int j = 0; j < 2; ++j)
        acc[i][j] = __builtin_amdgcn_wmma_f32_16x16x32_f16(
            false, af[i], false, bf[j], (short)0, acc[i][j], false, false);
    __syncthreads();
  }
#else
  const int lrow = tid >> 2;
  const int lseg = (tid & 3) * 8;
  for (int kt = 0; kt < nk; ++kt) {
    const int k0 = kt * BKK;
    for (int r = lrow; r < BM; r += 64) {
      *(v8h*)(&As[0][r * LDT + lseg]) =
          *(const v8h*)(A + (size_t)(m0 + r) * K + k0 + lseg);
      *(v8h*)(&Bs[0][r * LDT + lseg]) =
          *(const v8h*)(Bt + (size_t)(n0 + r) * K + k0 + lseg);
    }
    __syncthreads();
    const int lr = lane & 15;
    const int lh = (lane >> 4) * 8;
    v16h af[4], bf[2];
#pragma unroll
    for (int i = 0; i < 4; ++i)
      af[i] = frag_ld(&As[0][(wm + i * 16 + lr) * LDT + lh]);
#pragma unroll
    for (int j = 0; j < 2; ++j)
      bf[j] = frag_ld(&Bs[0][(wn + j * 16 + lr) * LDT + lh]);
#pragma unroll
    for (int i = 0; i < 4; ++i)
#pragma unroll
      for (int j = 0; j < 2; ++j)
        acc[i][j] = __builtin_amdgcn_wmma_f32_16x16x32_f16(
            false, af[i], false, bf[j], (short)0, acc[i][j], false, false);
    __syncthreads();
  }
#endif

  // C/D layout: VGPR r -> M = r (lanes 0-15) or r+8 (lanes 16-31); N = lane&15
  const int cn = lane & 15;
  const int cm = (lane >> 4) * 8;
#pragma unroll
  for (int i = 0; i < 4; ++i) {
#pragma unroll
    for (int j = 0; j < 2; ++j) {
      const int gn = n0 + wn + j * 16 + cn;
      const float bz = bias ? bias[gn] : 0.0f;
#pragma unroll
      for (int r = 0; r < 8; ++r) {
        const int gm = m0 + wm + i * 16 + cm + r;
        const size_t idx = (size_t)gm * N + gn;
        float v = acc[i][j][r] + bz;
        if (EPI == 1) v = 0.5f * v * (1.0f + erff(v * 0.70710678118654752f));
        if (EPI == 0 || EPI == 1) ((h16*)outp)[idx] = (h16)v;
        else if (EPI == 2)        ((float*)outp)[idx] = resid[idx] + v;
        else                      ((float*)outp)[idx] = v;
      }
    }
  }
}

// ---------------------------------------------------------------------------
// Weight prep: Wt[n*K+k] = (f16)W[k*N+n], tiled transpose. K,N multiples of 32.
// ---------------------------------------------------------------------------
__global__ __launch_bounds__(256) void wconv_k(
    const float* __restrict__ W, h16* __restrict__ Wt, int K, int N) {
  __shared__ float tile[32][33];
  const int kb = blockIdx.x * 32, nb = blockIdx.y * 32;
  const int tx = threadIdx.x & 31, ty = threadIdx.x >> 5;
  for (int r = ty; r < 32; r += 8)
    tile[r][tx] = W[(size_t)(kb + r) * N + nb + tx];
  __syncthreads();
  for (int r = ty; r < 32; r += 8)
    Wt[(size_t)(nb + r) * K + kb + tx] = (h16)tile[tx][r];
}

// ---------------------------------------------------------------------------
// LayerNorm over C cols, one block (256 thr) per row; optional residual add.
// ---------------------------------------------------------------------------
template<typename OutT>
__global__ __launch_bounds__(256) void ln_k(
    const float* __restrict__ in, const float* __restrict__ add,
    const float* __restrict__ g, const float* __restrict__ b,
    OutT* __restrict__ out, int C) {
  const int row = blockIdx.x, tid = threadIdx.x;
  __shared__ float red[256];
  __shared__ float s_mean, s_rstd;
  const size_t base = (size_t)row * C;
  float sum = 0.f, sq = 0.f;
  for (int c = tid; c < C; c += 256) {
    float v = in[base + c];
    if (add) v += add[base + c];
    sum += v; sq += v * v;
  }
  red[tid] = sum; __syncthreads();
  for (int s = 128; s > 0; s >>= 1) { if (tid < s) red[tid] += red[tid + s]; __syncthreads(); }
  if (tid == 0) s_mean = red[0] / (float)C;
  __syncthreads();
  red[tid] = sq; __syncthreads();
  for (int s = 128; s > 0; s >>= 1) { if (tid < s) red[tid] += red[tid + s]; __syncthreads(); }
  if (tid == 0) {
    float m = s_mean;
    s_rstd = rsqrtf(fmaxf(red[0] / (float)C - m * m, 0.f) + 1e-5f);
  }
  __syncthreads();
  const float m = s_mean, rs = s_rstd;
  for (int c = tid; c < C; c += 256) {
    float v = in[base + c];
    if (add) v += add[base + c];
    out[base + c] = (OutT)((v - m) * rs * g[c] + b[c]);
  }
}

__global__ void cast_k(const float* __restrict__ in, h16* __restrict__ out, int n) {
  for (int i = blockIdx.x * blockDim.x + threadIdx.x; i < n;
       i += gridDim.x * blockDim.x)
    out[i] = (h16)in[i];
}

// ---------------------------------------------------------------------------
// Fastformer combine: per (b,h) block, s=64, d=96.
// POOL=false: attn_out[b,s,h*96+d] = v_w[s]*v[s,d]           (f16)
// POOL=true : fused16[b, j*8+h] (j<96: q_glob, j>=96: v_glob) (f16)
//             resid[b, d*8+h] = q_global[d]                   (f32)
// ---------------------------------------------------------------------------
__device__ __forceinline__ void softmax64(float* arr, int tid) {
  __syncthreads();
  if (tid == 0) {
    float mx = -3.4e38f;
    for (int i = 0; i < 64; ++i) mx = fmaxf(mx, arr[i]);
    float s = 0.f;
    for (int i = 0; i < 64; ++i) { float e = expf(arr[i] - mx); arr[i] = e; s += e; }
    float inv = 1.f / s;
    for (int i = 0; i < 64; ++i) arr[i] *= inv;
  }
  __syncthreads();
}

template<bool POOL>
__global__ __launch_bounds__(128) void combine_k(
    const h16* __restrict__ q, const h16* __restrict__ k, const h16* __restrict__ v,
    const int* __restrict__ mask, h16* __restrict__ out16,
    float* __restrict__ residp) {
  const int b = blockIdx.x >> 3;
  const int h = blockIdx.x & 7;
  const int tid = threadIdx.x;
  __shared__ h16 qL[64][96], kL[64][96], vL[64][96];
  __shared__ float warr[64], garr[96], kg[96];
  const size_t base = (size_t)(b * 64) * 768 + (size_t)h * 96;
  for (int i = tid; i < 64 * 96; i += 128) {
    int s = i / 96, d = i % 96;
    size_t gi = base + (size_t)s * 768 + d;
    qL[s][d] = q[gi]; kL[s][d] = k[gi]; vL[s][d] = v[gi];
  }
  __syncthreads();
  if (tid < 64) {                       // softmax_s( sum_d q / sqrt(96) )
    float s = 0.f;
    for (int d = 0; d < 96; ++d) s += (float)qL[tid][d];
    warr[tid] = (mask[b * 64 + tid] != 0) ? s * 0.1020620726159658f : -10000.0f;
  }
  softmax64(warr, tid);
  if (tid < 96) {                       // q_global
    float gs = 0.f;
    for (int s = 0; s < 64; ++s) gs += warr[s] * (float)qL[s][tid];
    garr[tid] = gs;
  }
  __syncthreads();
  if (tid < 64) {                       // k logits = k . q_global
    float l = 0.f;
    for (int d = 0; d < 96; ++d) l += (float)kL[tid][d] * garr[d];
    warr[tid] = (mask[b * 64 + tid] != 0) ? l : -10000.0f;
  }
  softmax64(warr, tid);
  if (tid < 96) {                       // k_global
    float gs = 0.f;
    for (int s = 0; s < 64; ++s) gs += warr[s] * (float)kL[s][tid];
    kg[tid] = gs;
  }
  __syncthreads();
  if (tid < 64) {                       // v logits = q . k_global
    float l = 0.f;
    for (int d = 0; d < 96; ++d) l += (float)qL[tid][d] * kg[d];
    warr[tid] = (mask[b * 64 + tid] != 0) ? l : -10000.0f;
  }
  softmax64(warr, tid);
  if (!POOL) {
    for (int i = tid; i < 64 * 96; i += 128) {
      int s = i / 96, d = i % 96;
      out16[base + (size_t)s * 768 + d] = (h16)(warr[s] * (float)vL[s][d]);
    }
  } else {
    if (tid < 96) {                     // v_global (reuse kg)
      float gs = 0.f;
      for (int s = 0; s < 64; ++s) gs += warr[s] * (float)vL[s][tid];
      kg[tid] = gs;
    }
    __syncthreads();
    if (tid < 96) {
      out16[(size_t)b * 1536 + (size_t)tid * 8 + h]        = (h16)garr[tid];
      out16[(size_t)b * 1536 + (size_t)(96 + tid) * 8 + h] = (h16)kg[tid];
      residp[(size_t)b * 768 + (size_t)tid * 8 + h]        = garr[tid];
    }
  }
}

// ---------------------------------------------------------------------------
extern "C" void kernel_launch(void* const* d_in, const int* in_sizes, int n_in,
                              void* d_out, int out_size, void* d_ws, size_t ws_size,
                              hipStream_t stream) {
  (void)in_sizes; (void)n_in; (void)out_size; (void)ws_size;
  const int B = 1024, S = 64, HID = 768, FFNN = 3072, EOUT = 256;
  const int M = B * S;  // 65536

  const float* tok   = (const float*)d_in[0];
  const int*   mask  = (const int*)d_in[1];
  const float* n1g = (const float*)d_in[2],  * n1b = (const float*)d_in[3];
  const float* wq  = (const float*)d_in[4],  * bq  = (const float*)d_in[5];
  const float* wk  = (const float*)d_in[6],  * bk  = (const float*)d_in[7];
  const float* wv  = (const float*)d_in[8],  * bv  = (const float*)d_in[9];
  const float* wo  = (const float*)d_in[10], * bo  = (const float*)d_in[11];
  const float* n2g = (const float*)d_in[12], * n2b = (const float*)d_in[13];
  const float* w1  = (const float*)d_in[14], * b1  = (const float*)d_in[15];
  const float* w2  = (const float*)d_in[16], * b2  = (const float*)d_in[17];
  const float* pwq = (const float*)d_in[18], * pbq = (const float*)d_in[19];
  const float* pwk = (const float*)d_in[20], * pbk = (const float*)d_in[21];
  const float* pwv = (const float*)d_in[22], * pbv = (const float*)d_in[23];
  const float* pwo = (const float*)d_in[24], * pbo = (const float*)d_in[25];
  const float* png = (const float*)d_in[26], * pnb = (const float*)d_in[27];
  const float* tng = (const float*)d_in[28], * tnb = (const float*)d_in[29];
  const float* pjw = (const float*)d_in[30], * pjb = (const float*)d_in[31];
  const float* ong = (const float*)d_in[32], * onb = (const float*)d_in[33];

  // ---- workspace carving -------------------------------------------------
  char* ws = (char*)d_ws;
  size_t off = 0;
  auto carve = [&](size_t bytes) -> void* {
    void* p = ws + off;
    off = (off + bytes + 255) & ~(size_t)255;
    return p;
  };
  float* x    = (float*)carve((size_t)M * HID * 4);   // running residual
  h16*   a16  = (h16*)  carve((size_t)M * HID * 2);   // GEMM A staging
  h16*   q16  = (h16*)  carve((size_t)M * HID * 2);
  h16*   k16  = (h16*)  carve((size_t)M * HID * 2);
  h16*   v16  = (h16*)  carve((size_t)M * HID * 2);
  h16*   hbuf = (h16*)  carve((size_t)M * FFNN * 2);  // FFN intermediate
  h16*   wqT  = (h16*)  carve((size_t)HID * HID * 2);
  h16*   wkT  = (h16*)  carve((size_t)HID * HID * 2);
  h16*   wvT  = (h16*)  carve((size_t)HID * HID * 2);
  h16*   woT  = (h16*)  carve((size_t)HID * HID * 2);
  h16*   w1T  = (h16*)  carve((size_t)FFNN * HID * 2);
  h16*   w2T  = (h16*)  carve((size_t)HID * FFNN * 2);
  h16*   pwqT = (h16*)  carve((size_t)HID * HID * 2);
  h16*   pwkT = (h16*)  carve((size_t)HID * HID * 2);
  h16*   pwvT = (h16*)  carve((size_t)HID * HID * 2);
  h16*   pwoT = (h16*)  carve((size_t)HID * 2 * HID * 2);
  h16*   pjT  = (h16*)  carve((size_t)EOUT * HID * 2);
  h16*   fus  = (h16*)  carve((size_t)B * 2 * HID * 2);
  float* resp = (float*)carve((size_t)B * HID * 4);
  float* poolf= (float*)carve((size_t)B * HID * 4);
  h16*   pl16 = (h16*)  carve((size_t)B * HID * 2);
  float* news = (float*)carve((size_t)B * EOUT * 4);

  // x = copy of token_states (never mutate inputs)
  hipMemcpyAsync(x, tok, (size_t)M * HID * 4, hipMemcpyDeviceToDevice, stream);

  // ---- weight prep (transpose + f32->f16) --------------------------------
  auto wc = [&](const float* W, h16* Wt, int K, int N) {
    wconv_k<<<dim3(K / 32, N / 32), 256, 0, stream>>>(W, Wt, K, N);
  };
  wc(wq, wqT, HID, HID);  wc(wk, wkT, HID, HID);  wc(wv, wvT, HID, HID);
  wc(wo, woT, HID, HID);
  wc(w1, w1T, HID, FFNN); wc(w2, w2T, FFNN, HID);
  wc(pwq, pwqT, HID, HID); wc(pwk, pwkT, HID, HID); wc(pwv, pwvT, HID, HID);
  wc(pwo, pwoT, 2 * HID, HID);
  wc(pjw, pjT, HID, EOUT);

  const dim3 gBig(M / BM, HID / BN);     // 512 x 6
  const dim3 gFfn(M / BM, FFNN / BN);    // 512 x 24

  // ---- Fastformer block --------------------------------------------------
  ln_k<h16><<<M, 256, 0, stream>>>(x, nullptr, n1g, n1b, a16, HID);
  gemm16<0><<<gBig, 256, 0, stream>>>(a16, wqT, bq, q16, nullptr, M, HID, HID);
  gemm16<0><<<gBig, 256, 0, stream>>>(a16, wkT, bk, k16, nullptr, M, HID, HID);
  gemm16<0><<<gBig, 256, 0, stream>>>(a16, wvT, bv, v16, nullptr, M, HID, HID);
  combine_k<false><<<B * 8, 128, 0, stream>>>(q16, k16, v16, mask, a16, nullptr);
  gemm16<2><<<gBig, 256, 0, stream>>>(a16, woT, bo, x, x, M, HID, HID);  // x += attn@wo+bo

  ln_k<h16><<<M, 256, 0, stream>>>(x, nullptr, n2g, n2b, a16, HID);
  gemm16<1><<<gFfn, 256, 0, stream>>>(a16, w1T, b1, hbuf, nullptr, M, FFNN, HID);
  gemm16<2><<<gBig, 256, 0, stream>>>(hbuf, w2T, b2, x, x, M, HID, FFNN); // x += ffn

  // ---- Pooling + head ----------------------------------------------------
  cast_k<<<4096, 256, 0, stream>>>(x, a16, M * HID);
  gemm16<0><<<gBig, 256, 0, stream>>>(a16, pwqT, pbq, q16, nullptr, M, HID, HID);
  gemm16<0><<<gBig, 256, 0, stream>>>(a16, pwkT, pbk, k16, nullptr, M, HID, HID);
  gemm16<0><<<gBig, 256, 0, stream>>>(a16, pwvT, pbv, v16, nullptr, M, HID, HID);
  combine_k<true><<<B * 8, 128, 0, stream>>>(q16, k16, v16, mask, fus, resp);

  gemm16<3><<<dim3(B / BM, HID / BN), 256, 0, stream>>>(
      fus, pwoT, pbo, poolf, nullptr, B, HID, 2 * HID);
  ln_k<float><<<B, 256, 0, stream>>>(poolf, resp, png, pnb, poolf, HID); // LN(out+resid)
  ln_k<h16><<<B, 256, 0, stream>>>(poolf, nullptr, tng, tnb, pl16, HID); // token_norm
  gemm16<3><<<dim3(B / BM, EOUT / BN), 256, 0, stream>>>(
      pl16, pjT, pjb, news, nullptr, B, EOUT, HID);
  ln_k<float><<<B, 256, 0, stream>>>(news, nullptr, ong, onb, (float*)d_out, EOUT);
}
